// Encoder_62354335203969
// MI455X (gfx1250) — matrix operand order repaired and checked by reading, not compiled, not run on hardware
//
#include <hip/hip_runtime.h>
#include <hip/hip_bf16.h>

// CDNA5 / gfx1250 attention-LSTM. Wave32, WMMA bf16 16x16x32 with f32 accum.
// pre_x (67 MB) is t-invariant and L2-resident (192 MB L2). All B-operand
// weights are pre-swizzled into WMMA fragment-linear layout so every fragment
// load is 2x b128 (32 B per lane, contiguous). X is staged through LDS (bf16,
// transposed, padded stride) for the pre_x GEMM's A fragments.

typedef __attribute__((ext_vector_type(16))) __bf16 v16bf;
typedef __attribute__((ext_vector_type(8)))  float  v8f;

#define B_TOT  2048
#define T_LEN  128
#define N_IN   64
#define H_DIM  128
#define G_DIM  512   // 4*H
#define TILE_B 16
#define XT_STR (T_LEN + 8)   // padded LDS stride (mult. of 8 -> 16B aligned rows)

__device__ __forceinline__ float fast_tanh(float x) {
#if __has_builtin(__builtin_amdgcn_tanhf)
  return __builtin_amdgcn_tanhf(x);            // v_tanh_f32 on gfx1250
#elif __has_builtin(__builtin_amdgcn_tanh_f32)
  return __builtin_amdgcn_tanh_f32(x);
#else
  return 1.0f - 2.0f / (__expf(2.0f * x) + 1.0f);
#endif
}

__device__ __forceinline__ float fast_sigmoid(float x) {
  return 1.0f / (1.0f + __expf(-x));
}

__device__ __forceinline__ v8f wmma_bf16(v16bf a, v16bf b, v8f c) {
  return __builtin_amdgcn_wmma_f32_16x16x32_bf16(
      false, a, false, b, (short)0, c, false, false);
}

__device__ __forceinline__ v16bf as_v16bf(uint4 lo, uint4 hi) {
  union { uint4 u[2]; v16bf v; } cvt;
  cvt.u[0] = lo; cvt.u[1] = hi;
  return cvt.v;
}

// A-fragment (16x32 bf16) from row-major source, stride in elements.
// Lane's data = one contiguous 32B run: k in [16h,16h+16).
__device__ __forceinline__ v16bf load_A_frag_rm(const __bf16* base, int stride) {
  int lane = threadIdx.x & 31, half = lane >> 4, m = lane & 15;
  const __bf16* r = base + m * stride + 16 * half;
  return as_v16bf(*(const uint4*)r, *(const uint4*)(r + 8));
}

// B-fragment (32x16 bf16) from fragment-linear packed tile (1 KB/tile):
// element j of lane <-> (k = 16*(lane>>4) + j, n = lane&15).
__device__ __forceinline__ v16bf load_B_frag_packed(const __bf16* tile) {
  int lane = threadIdx.x & 31;
  const uint4* p = (const uint4*)tile + lane * 2;
  return as_v16bf(p[0], p[1]);
}

// ---- Kernel 0a: W_hh (512x128 f32) -> fragment-linear bf16 tiles ------------
// Tile (jt,kc): B(k=p,n=j) = W_hh[jt*16+n][kc*32+k]; tile index = jt*4+kc.
__global__ __launch_bounds__(256) void whh_pack_kernel(const float* __restrict__ Whh,
                                                       __bf16* __restrict__ WhhF) {
  int idx = blockIdx.x * 256 + threadIdx.x;          // 128 tiles * 512
  int tile = idx >> 9, w = idx & 511;
  int lane2 = w >> 4, j = w & 15;
  int kc = tile & 3, jt = tile >> 2;
  int p = kc * 32 + ((lane2 >> 4) << 4) + j;
  int jcol = jt * 16 + (lane2 & 15);
  WhhF[idx] = (__bf16)Whh[(size_t)jcol * H_DIM + p];
}

// ---- Kernel 0b: W1x (cols 256..383 of W_attn1) -> fragment-linear tiles -----
// Tile (ct,kc): B(k=t,n=col) = W1x[ct*16+n][kc*32+k]; tile index = ct*4+kc.
__global__ __launch_bounds__(256) void w1x_pack_kernel(const float* __restrict__ Wattn1,
                                                       __bf16* __restrict__ W1xF) {
  int idx = blockIdx.x * 256 + threadIdx.x;          // 32 tiles * 512
  int tile = idx >> 9, w = idx & 511;
  int lane2 = w >> 4, j = w & 15;
  int kc = tile & 3, ct = tile >> 2;
  int tt = kc * 32 + ((lane2 >> 4) << 4) + j;
  int col = ct * 16 + (lane2 & 15);
  W1xF[idx] = (__bf16)Wattn1[(size_t)col * 384 + 256 + tt];
}

// ---- Kernel 1: pre_x[b,n,k] = sum_t X[b,t,n]*W1x[k,t] + b_attn1[k] ----------
// One block per batch: stage X[b] through LDS (bf16, transposed [n][t]).
__global__ __launch_bounds__(256) void prex_kernel(const float* __restrict__ X,
                                                   const __bf16* __restrict__ W1xF,
                                                   const float* __restrict__ battn1,
                                                   float* __restrict__ preX) {
  __shared__ alignas(16) __bf16 sXT[N_IN][XT_STR];   // ~17 KB
  int tid = threadIdx.x;
  int b = blockIdx.x;
  const float* Xb = X + (size_t)b * T_LEN * N_IN;

  // coalesced float4 row loads -> transposed bf16 store
  for (int idx = tid; idx < T_LEN * N_IN / 4; idx += 256) {
    int t = idx >> 4;                 // 16 float4 per (t) row
    int n4 = (idx & 15) << 2;
    float4 v = ((const float4*)Xb)[idx];
    sXT[n4 + 0][t] = (__bf16)v.x;
    sXT[n4 + 1][t] = (__bf16)v.y;
    sXT[n4 + 2][t] = (__bf16)v.z;
    sXT[n4 + 3][t] = (__bf16)v.w;
  }
  __syncthreads();

  int wave = tid >> 5, lane = tid & 31, half = lane >> 4, nl = lane & 15;
  int mt = wave & 3;                  // n-row tile (0..3)
  int cq = wave >> 2;                 // column-tile group (0..1)

  v16bf a[4];
#pragma unroll
  for (int kc = 0; kc < 4; ++kc)
    a[kc] = load_A_frag_rm(&sXT[mt * 16][kc * 32], XT_STR);

#pragma unroll
  for (int q = 0; q < 4; ++q) {
    int ct = cq * 4 + q;
    v8f acc = {};
#pragma unroll
    for (int kc = 0; kc < 4; ++kc)
      acc = wmma_bf16(a[kc], load_B_frag_packed(W1xF + ((size_t)(ct * 4 + kc) << 9)), acc);
    int col = ct * 16 + nl;
    float bias = battn1[col];
#pragma unroll
    for (int i = 0; i < 8; ++i) {
      int r = mt * 16 + i + 8 * half;
      preX[((size_t)b * N_IN + r) * H_DIM + col] = acc[i] + bias;
    }
  }
}

// ---- Kernel 2: persistent recurrent scan, one WG per 16-batch tile ----------
__global__ __launch_bounds__(256) void recurrent_kernel(
    const float* __restrict__ X, const float* __restrict__ Wattn1,
    const float* __restrict__ w2g, const float* __restrict__ b2g,
    const float* __restrict__ Wih, const float* __restrict__ bih,
    const float* __restrict__ bhh, const float* __restrict__ preX,
    const __bf16* __restrict__ WhhF, float* __restrict__ out) {
  __shared__ alignas(16) __bf16 sW1h_f[32 * 512];   // tile = kc*8+jt   32 KB
  __shared__ alignas(16) __bf16 sW1s_f[32 * 512];   //                  32 KB
  __shared__ alignas(16) __bf16 sWih_f[64 * 512];   // tile = jt*2+kc   64 KB
  __shared__ alignas(16) __bf16 sH[TILE_B][H_DIM];  // h_{t-1}           4 KB
  __shared__ alignas(16) __bf16 sC[TILE_B][H_DIM];  // c_{t-1}           4 KB
  __shared__ alignas(16) float  sA[TILE_B][H_DIM];  // attn pre-act      8 KB
  __shared__ alignas(16) float  sE[TILE_B][N_IN];   // e scores          4 KB
  __shared__ alignas(16) __bf16 sXt[TILE_B][N_IN];  // alpha * x_t       2 KB
  __shared__ alignas(16) float  sW2[H_DIM];
  __shared__ alignas(16) float  sBl[G_DIM];
  // total ~152.5 KB LDS

  int tid = threadIdx.x;
  int wave = tid >> 5, lane = tid & 31, half = lane >> 4, nl = lane & 15;
  int b0 = blockIdx.x * TILE_B;

  // ---- one-time: pack weights into fragment-linear LDS tiles ----
  for (int idx = tid; idx < 32 * 512; idx += 256) {
    int tile = idx >> 9, w = idx & 511;
    int lane2 = w >> 4, j = w & 15;
    int kc = tile >> 3, jt = tile & 7;
    int p = kc * 32 + ((lane2 >> 4) << 4) + j;      // k of B = p index
    int n = jt * 16 + (lane2 & 15);                 // n of B = attn column
    sW1h_f[idx] = (__bf16)Wattn1[(size_t)n * 384 + p];
    sW1s_f[idx] = (__bf16)Wattn1[(size_t)n * 384 + H_DIM + p];
  }
  for (int idx = tid; idx < 64 * 512; idx += 256) {
    int tile = idx >> 9, w = idx & 511;
    int lane2 = w >> 4, j = w & 15;
    int kc = tile & 1, jt = tile >> 1;
    int k = kc * 32 + ((lane2 >> 4) << 4) + j;      // input-feature index
    int n = jt * 16 + (lane2 & 15);                 // gate column
    sWih_f[idx] = (__bf16)Wih[(size_t)n * N_IN + k];
  }
  for (int idx = tid; idx < G_DIM; idx += 256) sBl[idx] = bih[idx] + bhh[idx];
  if (tid < H_DIM) sW2[tid] = w2g[tid];
  for (int idx = tid; idx < TILE_B * H_DIM; idx += 256) {
    int bb = idx >> 7, k = idx & 127;
    sH[bb][k] = (__bf16)0.0f;
    sC[bb][k] = (__bf16)0.0f;
  }
  float b2 = b2g[0];
  v8f creg = {};   // this wave's 16-column f32 slice of c, VGPR-resident
  __syncthreads();

  for (int t = 0; t < T_LEN; ++t) {
    // h A-fragments: loaded once, reused by attention GEMM and gate GEMM
    v16bf aH[4];
#pragma unroll
    for (int kc = 0; kc < 4; ++kc)
      aH[kc] = load_A_frag_rm(&sH[0][kc * 32], H_DIM);

    // ---- a = h @ W1h^T + c @ W1s^T : wave owns attn columns [16w,16w+16) ----
    {
      v8f acc = {};
#pragma unroll
      for (int kc = 0; kc < 4; ++kc)
        acc = wmma_bf16(aH[kc], load_B_frag_packed(&sW1h_f[(kc * 8 + wave) << 9]), acc);
#pragma unroll
      for (int kc = 0; kc < 4; ++kc) {
        v16bf aC = load_A_frag_rm(&sC[0][kc * 32], H_DIM);
        acc = wmma_bf16(aC, load_B_frag_packed(&sW1s_f[(kc * 8 + wave) << 9]), acc);
      }
#pragma unroll
      for (int i = 0; i < 8; ++i)
        sA[i + 8 * half][wave * 16 + nl] = acc[i];
    }
    __syncthreads();

    // ---- e[b,n] = sum_k tanh(pre_x[b,n,k] + a[b,k]) * w2[k]  (L2-resident) --
#pragma unroll
    for (int q = 0; q < 4; ++q) {
      int idx = tid + q * 256;
      int bb = idx >> 6, n = idx & 63;
      const float4* px4 = (const float4*)(preX + ((size_t)(b0 + bb) * N_IN + n) * H_DIM);
      const float4* sa4 = (const float4*)&sA[bb][0];
      const float4* w24 = (const float4*)sW2;
      float s = 0.f;
#pragma unroll 4
      for (int k4 = 0; k4 < H_DIM / 4; ++k4) {
        float4 p = px4[k4], a4 = sa4[k4], w4 = w24[k4];
        s = __builtin_fmaf(fast_tanh(p.x + a4.x), w4.x, s);
        s = __builtin_fmaf(fast_tanh(p.y + a4.y), w4.y, s);
        s = __builtin_fmaf(fast_tanh(p.z + a4.z), w4.z, s);
        s = __builtin_fmaf(fast_tanh(p.w + a4.w), w4.w, s);
      }
      sE[bb][n] = s + b2;
    }
    __syncthreads();

    // ---- softmax over n (16 lanes per batch row, shfl_xor reductions),
    //      x_tilde = alpha * x_t ----
    {
      int bb = tid >> 4, ln = tid & 15;
      float v0 = sE[bb][ln +  0], v1 = sE[bb][ln + 16];
      float v2 = sE[bb][ln + 32], v3 = sE[bb][ln + 48];
      float m = fmaxf(fmaxf(v0, v1), fmaxf(v2, v3));
#pragma unroll
      for (int off = 1; off < 16; off <<= 1)
        m = fmaxf(m, __shfl_xor(m, off, 32));     // stays within 16-lane group
      float e0 = __expf(v0 - m), e1 = __expf(v1 - m);
      float e2 = __expf(v2 - m), e3 = __expf(v3 - m);
      float s = (e0 + e1) + (e2 + e3);
#pragma unroll
      for (int off = 1; off < 16; off <<= 1)
        s += __shfl_xor(s, off, 32);
      float inv = 1.0f / s;
      const float* xrow = X + ((size_t)(b0 + bb) * T_LEN + t) * N_IN;
      sXt[bb][ln +  0] = (__bf16)(e0 * inv * xrow[ln +  0]);
      sXt[bb][ln + 16] = (__bf16)(e1 * inv * xrow[ln + 16]);
      sXt[bb][ln + 32] = (__bf16)(e2 * inv * xrow[ln + 32]);
      sXt[bb][ln + 48] = (__bf16)(e3 * inv * xrow[ln + 48]);
    }
    __syncthreads();

    // ---- gates = x_tilde @ W_ih^T + h @ W_hh^T : wave owns j-cols
    //      [16w,16w+16) of each of the 4 gates ----
    v16bf aXt[2];
#pragma unroll
    for (int kc = 0; kc < 2; ++kc)
      aXt[kc] = load_A_frag_rm(&sXt[0][kc * 32], N_IN);

    v8f g4[4];
#pragma unroll
    for (int g = 0; g < 4; ++g) {
      int jt = g * 8 + wave;
      v8f acc = {};
#pragma unroll
      for (int kc = 0; kc < 2; ++kc)            // K = N_IN = 64
        acc = wmma_bf16(aXt[kc], load_B_frag_packed(&sWih_f[(jt * 2 + kc) << 9]), acc);
#pragma unroll
      for (int kc = 0; kc < 4; ++kc)            // K = H = 128, W_hh from L2
        acc = wmma_bf16(aH[kc], load_B_frag_packed(WhhF + ((size_t)(jt * 4 + kc) << 9)), acc);
      g4[g] = acc;
    }
    __syncthreads();   // all reads of sH/sC done before overwrite

    // ---- pointwise LSTM update; c stays in registers ----
    {
      int j = wave * 16 + nl;
      float bi = sBl[j], bfv = sBl[H_DIM + j];
      float bg = sBl[2 * H_DIM + j], bo = sBl[3 * H_DIM + j];
#pragma unroll
      for (int i = 0; i < 8; ++i) {
        int bb = i + 8 * half;
        float ig = fast_sigmoid(g4[0][i] + bi);
        float fg = fast_sigmoid(g4[1][i] + bfv);
        float gg = fast_tanh(g4[2][i] + bg);
        float og = fast_sigmoid(g4[3][i] + bo);
        float cn = fg * creg[i] + ig * gg;
        float hn = og * fast_tanh(cn);
        creg[i] = cn;
        sH[bb][j] = (__bf16)hn;
        sC[bb][j] = (__bf16)cn;
        out[((size_t)(b0 + bb) * T_LEN + t) * H_DIM + j] = hn;
      }
    }
    __syncthreads();
  }
}

extern "C" void kernel_launch(void* const* d_in, const int* in_sizes, int n_in,
                              void* d_out, int out_size, void* d_ws, size_t ws_size,
                              hipStream_t stream) {
  const float* X      = (const float*)d_in[0];
  const float* Wattn1 = (const float*)d_in[1];
  const float* battn1 = (const float*)d_in[2];
  const float* w2     = (const float*)d_in[3];
  const float* b2     = (const float*)d_in[4];
  const float* Wih    = (const float*)d_in[5];
  const float* Whh    = (const float*)d_in[6];
  const float* bih    = (const float*)d_in[7];
  const float* bhh    = (const float*)d_in[8];
  float* out = (float*)d_out;

  // workspace: [0, 64MB) pre_x f32; W_hh packed bf16 (128 KB); W1x packed (32 KB)
  char* ws = (char*)d_ws;
  float*  preX = (float*)ws;
  __bf16* WhhF = (__bf16*)(ws + (size_t)B_TOT * N_IN * H_DIM * sizeof(float));
  __bf16* W1xF = (__bf16*)(ws + (size_t)B_TOT * N_IN * H_DIM * sizeof(float)
                              + (size_t)G_DIM * H_DIM * sizeof(__bf16));

  whh_pack_kernel<<<(G_DIM * H_DIM) / 256, 256, 0, stream>>>(Whh, WhhF);
  w1x_pack_kernel<<<(H_DIM * H_DIM) / 256, 256, 0, stream>>>(Wattn1, W1xF);
  prex_kernel<<<B_TOT, 256, 0, stream>>>(X, W1xF, battn1, preX);
  recurrent_kernel<<<B_TOT / TILE_B, 256, 0, stream>>>(X, Wattn1, w2, b2, Wih,
                                                       bih, bhh, preX, WhhF, out);
}
